// PrimalPerturbationBlock_979252543699
// MI455X (gfx1250) — matrix operand order repaired and checked by reading, not compiled, not run on hardware
//
#include <hip/hip_runtime.h>
#include <hip/hip_bf16.h>

// ---------------------------------------------------------------------------
// PrimalPerturbationBlock for MI455X (gfx1250, wave32, WMMA).
// Dense matmuls: v_wmma_f32_16x16x32_bf16, LDS-staged tiles.
//   - weights staged TRANSPOSED in LDS so B fragments are K-contiguous
//     (ds_load_b128 pairs, no scalar packing)
//   - each wave: 16 rows x 32 cols -> A fragment reused for 2 WMMAs
//   - full-tile fast path: no per-element bounds checks (N % 64 == 0 here)
// Sparse phases (segment softmax, scatter-mean): float4 lanes + atomics.
// ---------------------------------------------------------------------------

typedef __attribute__((ext_vector_type(16))) __bf16 v16bf;
typedef __attribute__((ext_vector_type(8)))  __bf16 v8bf;
typedef __attribute__((ext_vector_type(8)))  float  v8f;

static __device__ inline __bf16 to_bf16(float f) {
  unsigned u = __float_as_uint(f);
  u += 0x7FFFu + ((u >> 16) & 1u);          // round-to-nearest-even
  unsigned short h = (unsigned short)(u >> 16);
  __bf16 b;
  __builtin_memcpy(&b, &h, 2);
  return b;
}

// ordered-uint encoding so unsigned atomicMax == float max
static __device__ inline unsigned enc_f32(float f) {
  unsigned u = __float_as_uint(f);
  return (u & 0x80000000u) ? ~u : (u | 0x80000000u);
}
static __device__ inline float dec_f32(unsigned e) {
  unsigned u = (e & 0x80000000u) ? (e & 0x7FFFFFFFu) : ~e;
  return __uint_as_float(u);
}
#define ENC_NEG_INF 0x007FFFFFu   // enc(-inf)

// ---------------------------------------------------------------------------
// Generic WMMA GEMM:  Y[N x 64] = act( [P0|P1|P2|P3] @ W + bias )
// Parts may carry a gather index (row -> g[row]).  K padded to KT*32.
// Block = 256 threads = 8 waves; block tile = 64 rows x 64 cols.
// Wave w: rows (w>>1)*16 .. +15, cols (w&1)*32 .. +31  (2 WMMAs / K-tile).
// ---------------------------------------------------------------------------
template<int KT>
__global__ __launch_bounds__(256)
void gemm_wmma_kernel(const float* __restrict__ p0, int w0, const int* __restrict__ g0,
                      const float* __restrict__ p1, int w1, const int* __restrict__ g1,
                      const float* __restrict__ p2, int w2, const int* __restrict__ g2,
                      const float* __restrict__ p3, int w3, const int* __restrict__ g3,
                      const float* __restrict__ Wm, const float* __restrict__ bias,
                      int Ktot, float* __restrict__ Y, int N, int do_relu)
{
  constexpr int KP = KT * 32;
  constexpr int KS = KP + 8;              // padded row stride (16B-aligned rows)
  __shared__ __bf16 lA[64 * KS];          // activations, row-major [r][k]
  __shared__ __bf16 lBt[64 * KS];         // weights TRANSPOSED:   [n][k]

  const int tid = threadIdx.x;
  const int rowbase = blockIdx.x * 64;
  const bool full = (rowbase + 64 <= N);  // N is a multiple of 64 in this model

  // stage weights transposed (read W[k*64+n] -> lBt[n][k]); zero-pad K
  for (int i = tid; i < 64 * KP; i += 256) {
    int n = i / KP, k = i - n * KP;
    float w = (k < Ktot) ? Wm[(size_t)k * 64 + n] : 0.f;
    lBt[n * KS + k] = to_bf16(w);
  }
  // stage activations (concat of parts, optional gather); zero-pad K (and rows
  // on the remainder path)
  if (full) {
    for (int i = tid; i < 64 * KP; i += 256) {
      int r = i / KP, k = i - r * KP;
      int row = rowbase + r;
      float x = 0.f;
      int kk = k;
      if (kk < w0)              x = p0[(size_t)(g0 ? g0[row] : row) * w0 + kk];
      else if ((kk -= w0) < w1) x = p1[(size_t)(g1 ? g1[row] : row) * w1 + kk];
      else if ((kk -= w1) < w2) x = p2[(size_t)(g2 ? g2[row] : row) * w2 + kk];
      else if ((kk -= w2) < w3) x = p3[(size_t)(g3 ? g3[row] : row) * w3 + kk];
      lA[r * KS + k] = to_bf16(x);
    }
  } else {
    for (int i = tid; i < 64 * KP; i += 256) {
      int r = i / KP, k = i - r * KP;
      int row = rowbase + r;
      float x = 0.f;
      if (row < N) {
        int kk = k;
        if (kk < w0)              x = p0[(size_t)(g0 ? g0[row] : row) * w0 + kk];
        else if ((kk -= w0) < w1) x = p1[(size_t)(g1 ? g1[row] : row) * w1 + kk];
        else if ((kk -= w1) < w2) x = p2[(size_t)(g2 ? g2[row] : row) * w2 + kk];
        else if ((kk -= w2) < w3) x = p3[(size_t)(g3 ? g3[row] : row) * w3 + kk];
      }
      lA[r * KS + k] = to_bf16(x);
    }
  }
  __syncthreads();

  const int wave = tid >> 5;
  const int lane = tid & 31;
  const int l15 = lane & 15;
  const int lh  = lane >> 4;
  const int rt  = (wave >> 1) * 16;       // row sub-tile of this wave
  const int ctb = (wave & 1) * 32;        // first of two 16-col tiles

  v8f acc0 = {0.f,0.f,0.f,0.f,0.f,0.f,0.f,0.f};
  v8f acc1 = {0.f,0.f,0.f,0.f,0.f,0.f,0.f,0.f};

  const int m = rt + l15;                 // A: lane holds row m (ISA 7.12.2)
  for (int kt = 0; kt < KT; ++kt) {
    // A fragment: K = lh*16 + [0..15]  (two b128 loads)
    const int ka = kt * 32 + lh * 16;
    v8bf a0 = *(const v8bf*)&lA[m * KS + ka];
    v8bf a1 = *(const v8bf*)&lA[m * KS + ka + 8];
    v16bf a = __builtin_shufflevector(a0, a1, 0,1,2,3,4,5,6,7,8,9,10,11,12,13,14,15);
    // B fragments: lane holds column n, K = lh*16 + [0..15] (K-contiguous)
    const int kbb = kt * 32 + lh * 16;
    {
      const int n = ctb + l15;
      v8bf b0 = *(const v8bf*)&lBt[n * KS + kbb];
      v8bf b1 = *(const v8bf*)&lBt[n * KS + kbb + 8];
      v16bf b = __builtin_shufflevector(b0, b1, 0,1,2,3,4,5,6,7,8,9,10,11,12,13,14,15);
      acc0 = __builtin_amdgcn_wmma_f32_16x16x32_bf16(false, a, false, b,
                                                     (short)0, acc0, false, false);
    }
    {
      const int n = ctb + 16 + l15;
      v8bf b0 = *(const v8bf*)&lBt[n * KS + kbb];
      v8bf b1 = *(const v8bf*)&lBt[n * KS + kbb + 8];
      v16bf b = __builtin_shufflevector(b0, b1, 0,1,2,3,4,5,6,7,8,9,10,11,12,13,14,15);
      acc1 = __builtin_amdgcn_wmma_f32_16x16x32_bf16(false, a, false, b,
                                                     (short)0, acc1, false, false);
    }
  }

  // C/D layout: VGPR v -> M = v + 8*(lane>>4), N = lane&15
  const int mrow = rowbase + rt + lh * 8;
#pragma unroll
  for (int t = 0; t < 2; ++t) {
    const v8f acc = t ? acc1 : acc0;
    const int col = ctb + t * 16 + l15;
    const float bv = bias ? bias[col] : 0.f;
    if (full) {
#pragma unroll
      for (int v = 0; v < 8; ++v) {
        float r = acc[v] + bv;
        if (do_relu) r = fmaxf(r, 0.f);
        Y[(size_t)(mrow + v) * 64 + col] = r;
      }
    } else {
#pragma unroll
      for (int v = 0; v < 8; ++v) {
        if (mrow + v < N) {
          float r = acc[v] + bv;
          if (do_relu) r = fmaxf(r, 0.f);
          Y[(size_t)(mrow + v) * 64 + col] = r;
        }
      }
    }
  }
}

// ---------------------------------------------------------------------------
// Sparse / elementwise kernels
// ---------------------------------------------------------------------------
__global__ void fill_f32_kernel(float* p, float v, long n) {
  long i = (long)blockIdx.x * blockDim.x + threadIdx.x;
  if (i < n) p[i] = v;
}
__global__ void fill_u32_kernel(unsigned* p, unsigned v, long n) {
  long i = (long)blockIdx.x * blockDim.x + threadIdx.x;
  if (i < n) p[i] = v;
}

// score[e] = dot(q[dst], k[src]+e[e]) / 8 ; segment max via encoded atomicMax
__global__ __launch_bounds__(256)
void edge_score_kernel(const float* __restrict__ q, const float* __restrict__ k,
                       const float* __restrict__ ee,
                       const int* __restrict__ src, const int* __restrict__ dst,
                       float* __restrict__ score, unsigned* __restrict__ menc, long NEdge)
{
  long eid = (long)blockIdx.x * 16 + (threadIdx.x >> 4);
  int l = threadIdx.x & 15;
  if (eid >= NEdge) return;
  int s = src[eid], d = dst[eid];
  const float4 qv = *(const float4*)(q  + (size_t)d   * 64 + l * 4);
  const float4 kv = *(const float4*)(k  + (size_t)s   * 64 + l * 4);
  const float4 ev = *(const float4*)(ee + (size_t)eid * 64 + l * 4);
  float p = qv.x*(kv.x+ev.x) + qv.y*(kv.y+ev.y) + qv.z*(kv.z+ev.z) + qv.w*(kv.w+ev.w);
  for (int off = 8; off; off >>= 1) p += __shfl_down(p, off, 16);
  if (l == 0) {
    float sc = p * 0.125f;                 // 1/sqrt(64)
    score[eid] = sc;
    atomicMax(menc + d, enc_f32(sc));
  }
}

__global__ void edge_expsum_kernel(const float* __restrict__ score,
                                   const unsigned* __restrict__ menc,
                                   const int* __restrict__ dst,
                                   float* __restrict__ ealpha,
                                   float* __restrict__ z, float* __restrict__ cnt,
                                   long NEdge)
{
  long e = (long)blockIdx.x * blockDim.x + threadIdx.x;
  if (e >= NEdge) return;
  int d = dst[e];
  float m = dec_f32(menc[d]);
  float a = expf(score[e] - m);
  ealpha[e] = a;
  atomicAdd(z + d, a);
  atomicAdd(cnt + d, 1.f);
}

// agg[dst] += alpha * (v[src] + e[e])
__global__ __launch_bounds__(256)
void edge_scatter_kernel(const float* __restrict__ ealpha, const float* __restrict__ z,
                         const float* __restrict__ vv, const float* __restrict__ ee,
                         const int* __restrict__ src, const int* __restrict__ dst,
                         float* __restrict__ agg, long NEdge)
{
  long eid = (long)blockIdx.x * 16 + (threadIdx.x >> 4);
  int l = threadIdx.x & 15;
  if (eid >= NEdge) return;
  int s = src[eid], d = dst[eid];
  float w = ealpha[eid] / z[d];
  const float4 v4 = *(const float4*)(vv + (size_t)s   * 64 + l * 4);
  const float4 e4 = *(const float4*)(ee + (size_t)eid * 64 + l * 4);
  float* ap = agg + (size_t)d * 64 + l * 4;
  atomicAdd(ap + 0, w * (v4.x + e4.x));
  atomicAdd(ap + 1, w * (v4.y + e4.y));
  atomicAdd(ap + 2, w * (v4.z + e4.z));
  atomicAdd(ap + 3, w * (v4.w + e4.w));
}

// out = relu(agg/max(cnt,1) + skip)
__global__ void node_finish_kernel(const float* __restrict__ agg, const float* __restrict__ cnt,
                                   const float* __restrict__ skip, float* __restrict__ out,
                                   long Nn)
{
  long i = (long)blockIdx.x * blockDim.x + threadIdx.x;
  if (i >= Nn * 64) return;
  long nid = i >> 6;
  float c = fmaxf(cnt[nid], 1.f);
  out[i] = fmaxf(agg[i] / c + skip[i], 0.f);
}

__global__ void mm_sign_kernel(const float* __restrict__ elp, const int* __restrict__ vidx,
                               const float* __restrict__ ts,
                               float* __restrict__ mms, float* __restrict__ mmc, long NEdge)
{
  long e = (long)blockIdx.x * blockDim.x + threadIdx.x;
  if (e >= NEdge) return;
  float t = tanhf(ts[0] * elp[e * 3 + 1]);   // PREV_MM_COL = 1
  int v = vidx[e];
  atomicAdd(mms + v, t);
  atomicAdd(mmc + v, 1.f);
}

__global__ void pert_final_kernel(const float* __restrict__ h2, const float* __restrict__ W3,
                                  const float* __restrict__ b3,
                                  const float* __restrict__ lpi, const float* __restrict__ msi,
                                  const float* __restrict__ mms, const float* __restrict__ mmc,
                                  float* __restrict__ plo, float* __restrict__ phi, int NVn)
{
  int v = blockIdx.x * blockDim.x + threadIdx.x;
  if (v >= NVn) return;
  float p = b3[0];
  const float* h = h2 + (size_t)v * 64;
#pragma unroll 8
  for (int d = 0; d < 64; ++d) p += h[d] * W3[d];
  float mm = mms[v] / fmaxf(mmc[v], 1.f);
  float pert = fabsf(lpi[0]) * p + fabsf(msi[0]) * mm;
  plo[v] = fmaxf(-pert + 0.0005f, 0.f);      // 0.5 * MIN_PERT
  phi[v] = fmaxf( pert + 0.0005f, 0.f);
}

__global__ void edge_cost_kernel(const float* __restrict__ plo, const float* __restrict__ phi,
                                 const float* __restrict__ dw, const int* __restrict__ vidx,
                                 float* __restrict__ olo, float* __restrict__ ohi, long NEdge)
{
  long e = (long)blockIdx.x * blockDim.x + threadIdx.x;
  if (e >= NEdge) return;
  int v = vidx[e];
  float w = dw[e];
  olo[e] = plo[v] * w;
  ohi[e] = phi[v] * w;
}

// ---------------------------------------------------------------------------
// Host side
// ---------------------------------------------------------------------------
extern "C" void kernel_launch(void* const* d_in, const int* in_sizes, int n_in,
                              void* d_out, int out_size, void* d_ws, size_t ws_size,
                              hipStream_t stream)
{
  (void)in_sizes; (void)n_in; (void)out_size; (void)ws_size;
  const int NV = 40000, NC = 16000, NE = 400000;

  const float* var_lp  = (const float*)d_in[0];
  const float* con_lp  = (const float*)d_in[1];
  const float* edge_lp = (const float*)d_in[2];
  const float* var_l   = (const float*)d_in[3];
  const float* con_l   = (const float*)d_in[4];
  const float* edge_l  = (const float*)d_in[5];
  const float* dw      = (const float*)d_in[6];
  const int*   eidx    = (const int*)  d_in[7];
  const int* vidx = eidx;
  const int* cidx = eidx + NE;

  // params in jax tree_flatten (sorted-key) order
  auto F = [&](int i){ return (const float*)d_in[i]; };
  const float *CC_We=F(8),*CC_Wk=F(9),*CC_Wq=F(10),*CC_Ws=F(11),*CC_Wv=F(12);
  const float *CC_bk=F(13),*CC_bq=F(14),*CC_bs=F(15),*CC_bv=F(16);
  const float *EU_W1=F(17),*EU_W2=F(18),*EU_b1=F(19),*EU_b2=F(20);
  const float *EUC_W1=F(21),*EUC_W2=F(22),*EUC_b1=F(23),*EUC_b2=F(24);
  const float *EUV_W1=F(25),*EUV_W2=F(26),*EUV_b1=F(27),*EUV_b2=F(28);
  const float *VC_We=F(29),*VC_Wk=F(30),*VC_Wq=F(31),*VC_Ws=F(32),*VC_Wv=F(33);
  const float *VC_bk=F(34),*VC_bq=F(35),*VC_bs=F(36),*VC_bv=F(37);
  const float *LPI=F(38),*MSI=F(39);
  const float *P_W1=F(40),*P_W2=F(41),*P_W3=F(42),*P_b1=F(43),*P_b2=F(44),*P_b3=F(45);
  const float *TS=F(46);

  float* out_vl  = (float*)d_out;
  float* out_cl  = out_vl + (size_t)NV * 64;
  float* out_el  = out_cl + (size_t)NC * 64;
  float* out_plo = out_el + (size_t)NE * 64;
  float* out_phi = out_plo + NE;

  float* wsf = (float*)d_ws;
  size_t off = 0;
  auto alloc = [&](size_t n){ float* p = wsf + off; off += n; return p; };
  float* eb  = alloc((size_t)NE * 64);   // e = edge_c @ We  (reused as edge MLP hidden)
  float* qb  = alloc((size_t)NV * 64);
  float* kb  = alloc((size_t)NV * 64);
  float* vb  = alloc((size_t)NV * 64);
  float* sb  = alloc((size_t)NV * 64);   // skip
  float* ab  = alloc((size_t)NV * 64);   // agg
  float* mb  = alloc(NV);                // encoded segment max (as unsigned)
  float* zb  = alloc(NV);
  float* cb  = alloc(NV);
  float* scb = alloc(NE);                // score
  float* alb = alloc(NE);                // exp(score - max)
  float* mms = alloc(NV);
  float* mmc = alloc(NV);
  float* plo = alloc(NV);
  float* phi = alloc(NV);

  auto gemm67 = [&](const float* x, const float* xlp, const float* W, const float* b,
                    float* y, int N, int relu) {
    gemm_wmma_kernel<3><<<dim3((N + 63) / 64), dim3(256), 0, stream>>>(
        x, 64, nullptr, xlp, 3, nullptr, nullptr, 0, nullptr, nullptr, 0, nullptr,
        W, b, 67, y, N, relu);
  };
  auto gemm64 = [&](const float* x, const float* W, const float* b,
                    float* y, int N, int relu) {
    gemm_wmma_kernel<2><<<dim3((N + 63) / 64), dim3(256), 0, stream>>>(
        x, 64, nullptr, nullptr, 0, nullptr, nullptr, 0, nullptr, nullptr, 0, nullptr,
        W, b, 64, y, N, relu);
  };

  auto tconv = [&](const float* xs_l, const float* xs_lp, int Ns,
                   const float* xd_l, const float* xd_lp, int Nd,
                   const int* sidx, const int* didx,
                   const float* We, const float* Wk, const float* Wq,
                   const float* Ws, const float* Wv,
                   const float* bk_, const float* bq_, const float* bs_, const float* bv_,
                   float* out) {
    gemm67(xd_l, xd_lp, Wq, bq_, qb, Nd, 0);
    gemm67(xs_l, xs_lp, Wk, bk_, kb, Ns, 0);
    gemm67(xs_l, xs_lp, Wv, bv_, vb, Ns, 0);
    gemm67(xd_l, xd_lp, Ws, bs_, sb, Nd, 0);
    gemm67(edge_l, edge_lp, We, nullptr, eb, NE, 0);
    fill_u32_kernel<<<(Nd + 255) / 256, 256, 0, stream>>>((unsigned*)mb, ENC_NEG_INF, Nd);
    fill_f32_kernel<<<(Nd + 255) / 256, 256, 0, stream>>>(zb, 0.f, Nd);
    fill_f32_kernel<<<(Nd + 255) / 256, 256, 0, stream>>>(cb, 0.f, Nd);
    fill_f32_kernel<<<(unsigned)(((long)Nd * 64 + 255) / 256), 256, 0, stream>>>(ab, 0.f, (long)Nd * 64);
    edge_score_kernel<<<(NE + 15) / 16, 256, 0, stream>>>(qb, kb, eb, sidx, didx, scb, (unsigned*)mb, NE);
    edge_expsum_kernel<<<(NE + 255) / 256, 256, 0, stream>>>(scb, (const unsigned*)mb, didx, alb, zb, cb, NE);
    edge_scatter_kernel<<<(NE + 15) / 16, 256, 0, stream>>>(alb, zb, vb, eb, sidx, didx, ab, NE);
    node_finish_kernel<<<(unsigned)(((long)Nd * 64 + 255) / 256), 256, 0, stream>>>(ab, cb, sb, out, Nd);
  };

  // --- stage 1: con update (messages var -> con) ---
  tconv(var_l, var_lp, NV, con_l, con_lp, NC, vidx, cidx,
        CC_We, CC_Wk, CC_Wq, CC_Ws, CC_Wv, CC_bk, CC_bq, CC_bs, CC_bv, out_cl);

  // --- stage 2: var update (messages con(updated) -> var) ---
  tconv(out_cl, con_lp, NC, var_l, var_lp, NV, cidx, vidx,
        VC_We, VC_Wk, VC_Wq, VC_Ws, VC_Wv, VC_bk, VC_bq, VC_bs, VC_bv, out_vl);

  // --- stage 3: edge update ---
  gemm67(out_vl, var_lp, EUV_W1, EUV_b1, kb, NV, 1);   // mlp2(var_c) hidden
  gemm64(kb, EUV_W2, EUV_b2, qb, NV, 1);               // hv -> qb
  gemm67(out_cl, con_lp, EUC_W1, EUC_b1, vb, NC, 1);   // mlp2(con_c) hidden
  gemm64(vb, EUC_W2, EUC_b2, sb, NC, 1);               // hc -> sb
  // cat = [edge_l(64) | edge_lp(3) | hv[vidx](64) | hc[cidx](64)] : K = 195
  gemm_wmma_kernel<7><<<dim3((NE + 63) / 64), dim3(256), 0, stream>>>(
      edge_l, 64, nullptr, edge_lp, 3, nullptr, qb, 64, vidx, sb, 64, cidx,
      EU_W1, EU_b1, 195, eb, NE, 1);
  gemm64(eb, EU_W2, EU_b2, out_el, NE, 1);

  // --- stage 4: mm_sign + perturbation head + per-edge costs ---
  fill_f32_kernel<<<(NV + 255) / 256, 256, 0, stream>>>(mms, 0.f, NV);
  fill_f32_kernel<<<(NV + 255) / 256, 256, 0, stream>>>(mmc, 0.f, NV);
  mm_sign_kernel<<<(NE + 255) / 256, 256, 0, stream>>>(edge_lp, vidx, TS, mms, mmc, NE);
  gemm64(out_vl, P_W1, P_b1, kb, NV, 1);
  gemm64(kb, P_W2, P_b2, qb, NV, 1);
  pert_final_kernel<<<(NV + 255) / 256, 256, 0, stream>>>(qb, P_W3, P_b3, LPI, MSI,
                                                          mms, mmc, plo, phi, NV);
  edge_cost_kernel<<<(NE + 255) / 256, 256, 0, stream>>>(plo, phi, dw, vidx,
                                                         out_plo, out_phi, NE);
}
